// OuterProductMean_50903952392542
// MI455X (gfx1250) — matrix operand order repaired and checked by reading, not compiled, not run on hardware
//
#include <hip/hip_runtime.h>

// ---------------------------------------------------------------------------
// OuterProductMean fused WMMA implementation for gfx1250 (MI455X, wave32).
//   S=128 seq, R=384 res, C_M=256, C=32, C_Z=128.
//   Stage A: transpose w_out -> bf16 wT[z][kappa'], kappa' = y*32+x.
//   Stage B: LayerNorm + proj -> aT[i][x][s], bT[j][y][s] (bf16, s-contig).
//   Stage C: per 4x4 (i,j) tile:
//       TDM tensor_load_to_lds stages 32KB a-slice + 32KB b-slice,
//       GEMM1 (outer product over s, fragments from LDS) -> packed O regs,
//       O -> LDS (b128 packed, y-major), GEMM2 (O @ wT) -> out.
//   Both GEMMs: v_wmma_f32_16x16x32_bf16.
// ---------------------------------------------------------------------------

#define R_RES 384
#define S_SEQ 128
#define C_M   256
#define C_HID 32
#define C_Z   128

typedef __attribute__((ext_vector_type(16))) __bf16 v16bf;
typedef __attribute__((ext_vector_type(8)))  float  v8f;
typedef __attribute__((ext_vector_type(4)))  unsigned int u32x4;
typedef __attribute__((ext_vector_type(8)))  int          i32x8;
typedef __attribute__((ext_vector_type(4)))  int          i32x4;

union FragCast {
    uint4 q[2];
    v16bf v;
};

static __device__ __forceinline__ unsigned short f32_to_bf16(float f) {
    union { float f; unsigned int u; } x;
    x.f = f;
    unsigned int r = x.u + 0x7FFFu + ((x.u >> 16) & 1u);   // RNE
    return (unsigned short)(r >> 16);
}

// Cheap pack: round-to-nearest (ties away) then one v_perm_b32 splice.
// 3 VALU per pair vs ~8 for the exact-RNE fallback.
static __device__ __forceinline__ unsigned int pack_bf16x2(float lo, float hi) {
    union { float f; unsigned int u; } a, b;
    a.f = lo; b.f = hi;
    unsigned int ra = a.u + 0x8000u;
    unsigned int rb = b.u + 0x8000u;
#if __has_builtin(__builtin_amdgcn_perm)
    // result bytes: {rb[3], rb[2], ra[3], ra[2]}  (upper halves of both)
    return __builtin_amdgcn_perm(rb, ra, 0x07060302u);
#else
    return (ra >> 16) | (rb & 0xFFFF0000u);
#endif
}

static __device__ __forceinline__ v16bf load_frag2(const unsigned short* lo,
                                                   const unsigned short* hi) {
    FragCast fc;
    fc.q[0] = *reinterpret_cast<const uint4*>(lo);
    fc.q[1] = *reinterpret_cast<const uint4*>(hi);
    return fc.v;
}

// Issue one TDM 1-row copy from global gaddr -> LDS byte offset laddr.
// Descriptor per CDNA5 ISA.  n_elem8 = count of 8-byte
// elements (data_size code 3).  6-arg builtin form (clang-23 / therock-10.0).
static __device__ __forceinline__ void tdm_load_1d(unsigned int laddr,
                                                   unsigned long long gaddr,
                                                   unsigned int n_elem8) {
    u32x4 g0;
    g0[0] = 1u;                                   // count=1 valid descriptor
    g0[1] = laddr;                                // lds_addr [63:32]
    g0[2] = (unsigned int)(gaddr & 0xFFFFFFFFull);        // global_addr lo
    g0[3] = (unsigned int)((gaddr >> 32) & 0x1FFFFFFull)  // global_addr[56:32]
            | (2u << 30);                         // type=2 ("image")
    i32x8 g1;
    g1[0] = (int)(3u << 16);                      // wg_mask=0, data_size=3 (8B)
    g1[1] = (int)((n_elem8 & 0xFFFFu) << 16);     // tensor_dim0[15:0]
    g1[2] = (int)(((n_elem8 >> 16) & 0xFFFFu)     // tensor_dim0[31:16]
            | (1u << 16));                        // tensor_dim1 = 1
    g1[3] = (int)((n_elem8 & 0xFFFFu) << 16);     // tile_dim0 = n_elem8
    g1[4] = 1;                                    // tile_dim1 = 1, tile_dim2 = 0
    g1[5] = (int)n_elem8;                         // tensor_dim0_stride lo
    g1[6] = (int)((n_elem8 & 0xFFFFu) << 16);     // dim0_stride hi=0, dim1_stride lo
    g1[7] = 0;
    i32x4 z4 = {0, 0, 0, 0};
    i32x8 z8 = {0, 0, 0, 0, 0, 0, 0, 0};
    __builtin_amdgcn_tensor_load_to_lds(g0, g1, z4, z4, z8, 0);
}

// ---------------------------------------------------------------------------
// Kernel 0: w_out (1024 x 128 f32, [kappa][z], kappa=x*32+y)
//           -> wT bf16 [z][kappa'], kappa' = y*32+x  (K-contiguous B layout)
// ---------------------------------------------------------------------------
__global__ void opm_transpose_wout(const float* __restrict__ w_out,
                                   unsigned short* __restrict__ wT) {
    int tid = blockIdx.x * 256 + threadIdx.x;      // 0 .. 131071
    int z  = tid >> 10;
    int kp = tid & 1023;                           // kappa' = y*32+x
    int y  = kp >> 5;
    int x  = kp & 31;
    wT[tid] = f32_to_bf16(w_out[(x * 32 + y) * C_Z + z]);
}

// ---------------------------------------------------------------------------
// Kernel 1: LayerNorm over C_M + two 256->32 projections.
// One wave32 per (s,i) row.  Block = 8 waves = 8 rows.
// ---------------------------------------------------------------------------
__global__ void opm_ln_proj(const float* __restrict__ m,
                            const float* __restrict__ ln_g,
                            const float* __restrict__ ln_b,
                            const float* __restrict__ w1,
                            const float* __restrict__ b1,
                            const float* __restrict__ w2,
                            const float* __restrict__ b2,
                            unsigned short* __restrict__ aT,
                            unsigned short* __restrict__ bT) {
    __shared__ float mnbuf[8][C_M];

    const int lane = threadIdx.x & 31;
    const int wv   = threadIdx.x >> 5;
    const int row  = blockIdx.x * 8 + wv;          // 0 .. S*R-1
    const int s    = row / R_RES;
    const int i    = row % R_RES;

    const float* mp = m + (size_t)row * C_M + lane * 8;
    float4 v0 = reinterpret_cast<const float4*>(mp)[0];
    float4 v1 = reinterpret_cast<const float4*>(mp)[1];
    float e[8] = { v0.x, v0.y, v0.z, v0.w, v1.x, v1.y, v1.z, v1.w };

    float sum = 0.f, sq = 0.f;
#pragma unroll
    for (int t = 0; t < 8; ++t) { sum += e[t]; sq += e[t] * e[t]; }
#pragma unroll
    for (int off = 16; off > 0; off >>= 1) {
        sum += __shfl_xor(sum, off, 32);
        sq  += __shfl_xor(sq,  off, 32);
    }
    const float mu  = sum * (1.0f / C_M);
    const float var = sq * (1.0f / C_M) - mu * mu;
    const float rs  = rsqrtf(var + 1e-5f);

#pragma unroll
    for (int t = 0; t < 8; ++t) {
        const int k = lane * 8 + t;
        mnbuf[wv][k] = (e[t] - mu) * rs * ln_g[k] + ln_b[k];
    }
    __syncthreads();

    float aacc = b1[lane];
    float bacc = b2[lane];
#pragma unroll 8
    for (int k = 0; k < C_M; ++k) {
        const float mk = mnbuf[wv][k];
        aacc = fmaf(mk, w1[k * C_HID + lane], aacc);
        bacc = fmaf(mk, w2[k * C_HID + lane], bacc);
    }

    const size_t o = ((size_t)i * C_HID + lane) * S_SEQ + s;
    aT[o] = f32_to_bf16(aacc);
    bT[o] = f32_to_bf16(bacc);
}

// ---------------------------------------------------------------------------
// Kernel 2: fused outer-product (over s) + output projection.
// Block = 256 threads (8 waves), owns 4 i-rows x 4 j-rows = 16 (i,j) pairs.
// LDS (64KB) is time-shared: [stage a 32KB | stage b 32KB] then [O 32KB].
// ---------------------------------------------------------------------------
__global__ void opm_fused(const unsigned short* __restrict__ aT,
                          const unsigned short* __restrict__ bT,
                          const unsigned short* __restrict__ wT,
                          const float* __restrict__ b_out,
                          float* __restrict__ out) {
    __shared__ __align__(16) unsigned char lds_raw[65536];

    unsigned short* stageA = reinterpret_cast<unsigned short*>(lds_raw);
    unsigned short* stageB = stageA + 4 * C_HID * S_SEQ;      // +16384 halfs
    unsigned short* O_lds  = reinterpret_cast<unsigned short*>(lds_raw);

    const int lane = threadIdx.x & 31;
    const int wv   = threadIdx.x >> 5;
    const int half = lane >> 4;
    const int ln   = lane & 15;

    const int i0 = blockIdx.y * 4;
    const int j0 = blockIdx.x * 4;

    // ---- Phase 0: TDM-stage the 32KB a-slice and 32KB b-slice into LDS ----
    if (wv == 0) {
        const unsigned int ldsA = (unsigned int)(uintptr_t)(void*)stageA;
        const unsigned int ldsB = (unsigned int)(uintptr_t)(void*)stageB;
        const unsigned long long ga =
            (unsigned long long)(uintptr_t)(aT + (size_t)i0 * C_HID * S_SEQ);
        const unsigned long long gb =
            (unsigned long long)(uintptr_t)(bT + (size_t)j0 * C_HID * S_SEQ);
        tdm_load_1d(ldsA, ga, 4096);   // 4096 x 8B = 32KB
        tdm_load_1d(ldsB, gb, 4096);
        __builtin_amdgcn_s_wait_tensorcnt((short)0);
    }
    __syncthreads();

    // ---- Phase 1: GEMM1, fragments from LDS; pack O into registers ----
    uint4 opack[2][4];                 // [pair][tile xt*2+yt], bf16-packed

#pragma unroll
    for (int pp = 0; pp < 2; ++pp) {
        const int p  = wv * 2 + pp;    // pair index 0..15
        const int di = p >> 2;
        const int dj = p & 3;

        const unsigned short* ab = stageA + di * C_HID * S_SEQ;
        const unsigned short* bb = stageB + dj * C_HID * S_SEQ;

        v8f acc[2][2];
#pragma unroll
        for (int xt = 0; xt < 2; ++xt)
#pragma unroll
            for (int yt = 0; yt < 2; ++yt)
                acc[xt][yt] = (v8f){0,0,0,0,0,0,0,0};

#pragma unroll
        for (int k = 0; k < 4; ++k) {
            const int s0 = k * 32;
            v16bf af[2], bf[2];
#pragma unroll
            for (int t = 0; t < 2; ++t) {
                const unsigned short* pa = ab + (t * 16 + ln) * S_SEQ + s0 + half * 8;
                af[t] = load_frag2(pa, pa + 16);
                const unsigned short* pb = bb + (t * 16 + ln) * S_SEQ + s0 + half * 8;
                bf[t] = load_frag2(pb, pb + 16);
            }
#pragma unroll
            for (int xt = 0; xt < 2; ++xt)
#pragma unroll
                for (int yt = 0; yt < 2; ++yt)
                    acc[xt][yt] = __builtin_amdgcn_wmma_f32_16x16x32_bf16(
                        false, af[xt], false, bf[yt], (short)0, acc[xt][yt],
                        false, false);
        }

#pragma unroll
        for (int xt = 0; xt < 2; ++xt)
#pragma unroll
            for (int yt = 0; yt < 2; ++yt) {
                uint4 pk;
                pk.x = pack_bf16x2(acc[xt][yt][0], acc[xt][yt][1]);
                pk.y = pack_bf16x2(acc[xt][yt][2], acc[xt][yt][3]);
                pk.z = pack_bf16x2(acc[xt][yt][4], acc[xt][yt][5]);
                pk.w = pack_bf16x2(acc[xt][yt][6], acc[xt][yt][7]);
                opack[pp][xt * 2 + yt] = pk;
            }
    }
    __syncthreads();   // everyone done reading the stage -> LDS reusable

    // ---- Phase 2: store O (y-major: O[p][y*32+x]) as packed b128 ----
#pragma unroll
    for (int pp = 0; pp < 2; ++pp) {
        const int p = wv * 2 + pp;
        unsigned short* od = &O_lds[p * 1024];
#pragma unroll
        for (int xt = 0; xt < 2; ++xt)
#pragma unroll
            for (int yt = 0; yt < 2; ++yt) {
                // lane ln holds y = yt*16+ln, x-run = xt*16 + half*8 .. +8
                unsigned short* dst = od + (yt * 16 + ln) * 32 + xt * 16 + half * 8;
                *reinterpret_cast<uint4*>(dst) = opack[pp][xt * 2 + yt];
            }
    }
    __syncthreads();

    // ---- Phase 3: GEMM2  (16 pairs x 1024) @ (1024 x 128) ----
    const int z = wv * 16 + ln;        // wave owns z tile [wv*16, wv*16+16)
    v8f acc2 = {0,0,0,0,0,0,0,0};

#pragma unroll 4
    for (int k = 0; k < 32; ++k) {
        const int k0 = k * 32;
        const unsigned short* pa = &O_lds[ln * 1024 + k0 + half * 8];
        v16bf af = load_frag2(pa, pa + 16);
        const unsigned short* pb = wT + (size_t)z * 1024 + k0 + half * 8;
        v16bf bf = load_frag2(pb, pb + 16);
        acc2 = __builtin_amdgcn_wmma_f32_16x16x32_bf16(false, af, false, bf,
                                                       (short)0, acc2, false, false);
    }

    const float bz = b_out[z];
#pragma unroll
    for (int r = 0; r < 8; ++r) {
        const int p  = half * 8 + r;
        const int di = p >> 2;
        const int dj = p & 3;
        const size_t o = ((size_t)(i0 + di) * R_RES + (j0 + dj)) * C_Z + z;
        out[o] = (acc2[r] + bz) * (1.0f / S_SEQ);
    }
}

// ---------------------------------------------------------------------------
// Launch: inputs 0:m 1:ln_g 2:ln_b 3:w1 4:b1 5:w2 6:b2 7:w_out 8:b_out
// Workspace (halfs): aT[384*32*128] | bT[384*32*128] | wT[128*1024] = 6.5MB
// ---------------------------------------------------------------------------
extern "C" void kernel_launch(void* const* d_in, const int* in_sizes, int n_in,
                              void* d_out, int out_size, void* d_ws, size_t ws_size,
                              hipStream_t stream) {
    const float* m     = (const float*)d_in[0];
    const float* ln_g  = (const float*)d_in[1];
    const float* ln_b  = (const float*)d_in[2];
    const float* w1    = (const float*)d_in[3];
    const float* b1    = (const float*)d_in[4];
    const float* w2    = (const float*)d_in[5];
    const float* b2    = (const float*)d_in[6];
    const float* w_out = (const float*)d_in[7];
    const float* b_out = (const float*)d_in[8];
    float* out = (float*)d_out;

    unsigned short* aT = (unsigned short*)d_ws;
    unsigned short* bT = aT + (size_t)R_RES * C_HID * S_SEQ;
    unsigned short* wT = bT + (size_t)R_RES * C_HID * S_SEQ;

    opm_transpose_wout<<<(1024 * 128) / 256, 256, 0, stream>>>(w_out, wT);

    opm_ln_proj<<<(S_SEQ * R_RES) / 8, 256, 0, stream>>>(
        m, ln_g, ln_b, w1, b1, w2, b2, aT, bT);

    dim3 grid(R_RES / 4, R_RES / 4);   // 96 x 96 blocks, 16 (i,j) pairs each
    opm_fused<<<grid, 256, 0, stream>>>(aT, bT, wT, b_out, out);
}
